// NACE_27874337751540
// MI455X (gfx1250) — matrix operand order repaired and checked by reading, not compiled, not run on hardware
//
#include <hip/hip_runtime.h>

typedef unsigned short ushort_t;
typedef unsigned int   uint_t;

typedef __attribute__((ext_vector_type(16))) __bf16 v16bf;
typedef __attribute__((ext_vector_type(8)))  float  v8f;

union Frag { v16bf v; uint_t d[8]; };

// Native bf16 convert (backend picks v_cvt*bf16* on gfx1250), packed pair.
__device__ __forceinline__ uint_t pack2bf(float a, float b) {
  union { __bf16 h[2]; uint_t u; } r;
  r.h[0] = (__bf16)a;
  r.h[1] = (__bf16)b;
  return r.u;
}

// ---- CDNA5 async global->LDS copy path (ASYNCcnt-tracked) -----------------
#if defined(__gfx1250__) && __has_builtin(__builtin_amdgcn_global_load_async_to_lds_b128)
#define NACE_ASYNC 1
typedef int v4i_t __attribute__((vector_size(16)));
typedef __attribute__((address_space(1))) v4i_t* gv4i_p;   // global (AS1) int4*
typedef __attribute__((address_space(3))) v4i_t* lv4i_p;   // LDS    (AS3) int4*
__device__ __forceinline__ void async_cp16(const void* g, void* l) {
  __builtin_amdgcn_global_load_async_to_lds_b128((gv4i_p)g, (lv4i_p)l, 0, 0);
}
__device__ __forceinline__ void async_wait0() {
#if __has_builtin(__builtin_amdgcn_s_wait_asynccnt)
  __builtin_amdgcn_s_wait_asynccnt(0);
#else
  asm volatile("s_wait_asynccnt 0" ::: "memory");
#endif
}
#else
#define NACE_ASYNC 0
#endif

// ---------------------------------------------------------------------------
// Kernel 1: per-(b,c) spatial mean of state -> means[128]
// ---------------------------------------------------------------------------
__global__ __launch_bounds__(256) void nace_mean_kernel(
    const float* __restrict__ state, float* __restrict__ means) {
  __shared__ float red[256];
  const int bc = blockIdx.x;                         // 0..127
  const float* p = state + ((size_t)bc << 16);       // 65536 elems
  float s = 0.f;
  for (int i = threadIdx.x; i < 65536; i += 256) s += p[i];
  red[threadIdx.x] = s;
  __syncthreads();
  for (int off = 128; off > 0; off >>= 1) {
    if (threadIdx.x < off) red[threadIdx.x] += red[threadIdx.x + off];
    __syncthreads();
  }
  if (threadIdx.x == 0) means[bc] = red[0] * (1.0f / 65536.0f);
}

// ---------------------------------------------------------------------------
// Kernel 2: weights fp32 -> bf16 pairs (W1 padded K:108->128)
// ---------------------------------------------------------------------------
__global__ __launch_bounds__(256) void nace_wprep_kernel(
    const float* __restrict__ W1, const float* __restrict__ W2,
    uint_t* __restrict__ W1bf, uint_t* __restrict__ W2bf) {
  int idx = blockIdx.x * 256 + threadIdx.x;          // dword (bf16-pair) index
  if (idx < 256 * 64) {                              // W1: 256 rows x 64 dwords
    int r = idx >> 6, k = (idx & 63) << 1;
    float v0 = (k < 108) ? W1[r * 108 + k] : 0.f;
    float v1 = (k + 1 < 108) ? W1[r * 108 + k + 1] : 0.f;
    W1bf[idx] = pack2bf(v0, v1);
  } else {
    int j = idx - 256 * 64;                          // W2: 2048 dwords
    if (j < 16 * 128) W2bf[j] = pack2bf(W2[2 * j], W2[2 * j + 1]);
  }
}

// ---------------------------------------------------------------------------
// Feature value for input channel k of one pixel.
// ---------------------------------------------------------------------------
__device__ __forceinline__ float feat_val(
    int k, int bIdx, int rem, int y, int x,
    const float* __restrict__ state, const float* __restrict__ action,
    const float* __restrict__ extra, const float* __restrict__ means) {
  float v = 0.f;
  if (k < 80) {                                      // perception taps, ch = c*5+tap
    int c = k / 5, tap = k - 5 * c;
    int dy = (tap == 1) ? -1 : (tap == 2) ? 1 : 0;
    int dx = (tap == 3) ? -1 : (tap == 4) ? 1 : 0;
    int yy = y + dy, xx = x + dx;
    if (yy >= 0 && yy < 256 && xx >= 0 && xx < 256)
      v = state[(((size_t)bIdx * 16 + c) << 16) + (yy << 8) + xx];
  } else if (k < 96) {                               // global mean context
    v = means[bIdx * 16 + (k - 80)];
  } else if (k < 104) {                              // action map
    v = action[(((size_t)bIdx * 8 + (k - 96)) << 16) + rem];
  } else if (k < 108) {                              // extra map
    v = extra[(((size_t)bIdx * 4 + (k - 104)) << 16) + rem];
  }                                                  // [108,128): zero pad
  return v;
}

// ---------------------------------------------------------------------------
// Kernel 3: main NCA step.  Wave = 16 pixels.  Block = 8 waves = 128 pixels.
// GEMM1: h[256 x 16px] = W1[256x128] * feat[128x16px]  (16 Mtiles x 4 Ksteps)
// GEMM2: dx[16 x 16px] = W2[16x256]  * h[256x16px]     (8 Ksteps)
// ---------------------------------------------------------------------------
__global__ __launch_bounds__(256) void nace_main_kernel(
    const float* __restrict__ state, const float* __restrict__ action,
    const float* __restrict__ extra, const float* __restrict__ b1,
    const float* __restrict__ b2, const float* __restrict__ means,
    const ushort_t* __restrict__ W1bf, const ushort_t* __restrict__ W2bf,
    float* __restrict__ out) {
  __shared__ __align__(16) ushort_t sW1[256 * 128];     // 64 KB bf16 W1 (padded)
  __shared__ __align__(16) ushort_t sW2[16 * 256];      //  8 KB bf16 W2
  __shared__ __align__(16) ushort_t sPool[8][16 * 256]; // 64 KB: feat then h
  __shared__ float sB1[256];
  __shared__ float sB2[16];

  const int tid  = threadIdx.x;
  const int lane = tid & 31;
  const int wave = tid >> 5;
  const int m    = lane & 15;                // column (pixel) / M-row within tile
  const int half = lane >> 4;                // 0: lanes 0-15, 1: lanes 16-31

  // ---- stage weights into LDS (async DMA, overlapped with feature build) --
#if NACE_ASYNC
  {
    const char* g1 = (const char*)W1bf;
    char*       l1 = (char*)sW1;
#pragma unroll
    for (int i = 0; i < 16; ++i) {                      // 65536 B
      int off = (tid + i * 256) * 16;
      async_cp16(g1 + off, l1 + off);
    }
    const char* g2 = (const char*)W2bf;
    char*       l2 = (char*)sW2;
    async_cp16(g2 + tid * 16, l2 + tid * 16);           // 8192 B
    async_cp16(g2 + (tid + 256) * 16, l2 + (tid + 256) * 16);
  }
#else
  {
    const uint_t* s1 = (const uint_t*)W1bf;
    uint_t*       d1 = (uint_t*)sW1;
    for (int i = tid; i < 256 * 128 / 2; i += 256) d1[i] = s1[i];
    const uint_t* s2 = (const uint_t*)W2bf;
    uint_t*       d2 = (uint_t*)sW2;
    for (int i = tid; i < 16 * 256 / 2; i += 256) d2[i] = s2[i];
  }
#endif
  sB1[tid] = b1[tid];
  if (tid < 16) sB2[tid] = b2[tid];

  // ---- build bf16 feature tile [16 px][128 K] in this wave's pool ---------
  const int pixBase = blockIdx.x * 128 + wave * 16;
  const int bIdx    = pixBase >> 16;                 // batch (block never straddles)
  ushort_t* feat  = sPool[wave];
  uint_t*   featd = (uint_t*)feat;
  for (int i2 = lane; i2 < 16 * 64; i2 += 32) {      // bf16-pair granularity
    int pix = i2 >> 6;
    int k   = (i2 & 63) << 1;
    int rem = (pixBase + pix) & 65535;
    int y = rem >> 8, x = rem & 255;
    float v0 = feat_val(k,     bIdx, rem, y, x, state, action, extra, means);
    float v1 = feat_val(k + 1, bIdx, rem, y, x, state, action, extra, means);
    featd[i2] = pack2bf(v0, v1);
  }
#if NACE_ASYNC
  async_wait0();                                     // weights landed in LDS
#endif
  __syncthreads();                                   // publish sW1/sW2/sB*

  // ---- B fragments (feat) for all 4 K-steps, kept in registers ------------
  // B 32x16 in this code's contiguous K order: lane col = N, VGPR r holds K = half*16 + 2r, +1
  Frag bf1[4];
  {
    const uint_t* fp = (const uint_t*)feat;          // [16 px][64 dwords]
#pragma unroll
    for (int t = 0; t < 4; ++t)
#pragma unroll
      for (int r = 0; r < 8; ++r)
        bf1[t].d[r] = fp[m * 64 + t * 16 + half * 8 + r];
  }

  // ---- layer 1: 16 M-tiles x 4 K-steps of v_wmma_f32_16x16x32_bf16 --------
  // A 16x32 in this code's contiguous K order (same as B): lane row = M, VGPR r holds K = half*16 + 2r, +1
  ushort_t* hbuf = sPool[wave];                      // reuse pool: h[16 px][256]
  uint_t*   hrow = (uint_t*)hbuf + m * 128;          // this lane's pixel row
  const uint_t* wp = (const uint_t*)sW1;             // [256][64 dwords]
#pragma unroll 1
  for (int ct = 0; ct < 16; ++ct) {
    const int mBase = ct * 16;
    v8f acc = {};
#pragma unroll
    for (int t = 0; t < 4; ++t) {
      Frag a;
#pragma unroll
      for (int r = 0; r < 8; ++r)
        a.d[r] = wp[(mBase + m) * 64 + t * 16 + half * 8 + r];
      acc = __builtin_amdgcn_wmma_f32_16x16x32_bf16(
          false, a.v, false, bf1[t].v, (short)0, acc, false, false);
    }
    // D layout: col = m (pixel), row = mBase + r + half*8.  Bias + LeakyReLU.
#pragma unroll
    for (int j = 0; j < 4; ++j) {
      int hc = mBase + half * 8 + 2 * j;
      float h0 = acc[2 * j] + sB1[hc];
      float h1 = acc[2 * j + 1] + sB1[hc + 1];
      h0 = h0 > 0.f ? h0 : 0.01f * h0;
      h1 = h1 > 0.f ? h1 : 0.01f * h1;
      hrow[hc >> 1] = pack2bf(h0, h1);
    }
  }

  // ---- layer 2: M=16, K=256 -> 8 K-steps ----------------------------------
  v8f acc2 = {};
  const uint_t* hp  = (const uint_t*)hbuf;           // [16 px][128 dwords]
  const uint_t* w2p = (const uint_t*)sW2;            // [16][128 dwords]
#pragma unroll 1
  for (int t = 0; t < 8; ++t) {
    Frag a, bb;
#pragma unroll
    for (int r = 0; r < 8; ++r) {
      a.d[r]  = w2p[m * 128 + t * 16 + half * 8 + r];
      bb.d[r] = hp[m * 128 + t * 16 + half * 8 + r];
    }
    acc2 = __builtin_amdgcn_wmma_f32_16x16x32_bf16(
        false, a.v, false, bb.v, (short)0, acc2, false, false);
  }

  // ---- epilogue: hardtanh(-0.5,0.5) + residual ----------------------------
  const int rem = (pixBase + m) & 65535;
#pragma unroll
  for (int r = 0; r < 8; ++r) {
    int co = r + half * 8;
    float dv = acc2[r] + sB2[co];
    dv = fminf(0.5f, fmaxf(-0.5f, dv));
    size_t gi = (((size_t)bIdx * 16 + co) << 16) + rem;
    out[gi] = state[gi] + dv;
  }
}

// ---------------------------------------------------------------------------
extern "C" void kernel_launch(void* const* d_in, const int* in_sizes, int n_in,
                              void* d_out, int out_size, void* d_ws, size_t ws_size,
                              hipStream_t stream) {
  const float* state  = (const float*)d_in[0];
  const float* action = (const float*)d_in[1];
  const float* extra  = (const float*)d_in[2];
  const float* W1     = (const float*)d_in[3];
  const float* b1     = (const float*)d_in[4];
  const float* W2     = (const float*)d_in[5];
  const float* b2     = (const float*)d_in[6];
  float* out = (float*)d_out;

  float*    means = (float*)d_ws;                               // 128 f32
  ushort_t* W1bf  = (ushort_t*)((char*)d_ws + 512);             // 256*128 bf16
  ushort_t* W2bf  = (ushort_t*)((char*)d_ws + 512 + 65536);     // 16*256 bf16

  nace_mean_kernel<<<128, 256, 0, stream>>>(state, means);
  nace_wprep_kernel<<<(256 * 64 + 16 * 128 + 255) / 256, 256, 0, stream>>>(
      W1, W2, (uint_t*)W1bf, (uint_t*)W2bf);
  nace_main_kernel<<<4096, 256, 0, stream>>>(state, action, extra, b1, b2,
                                             means, W1bf, W2bf, out);
}
